// MixGRU_17626545783147
// MI455X (gfx1250) — matrix operand order repaired and checked by reading, not compiled, not run on hardware
//
#include <hip/hip_runtime.h>

#define IN_D 512
#define MIXD 32
#define HID  96
#define G3   288   // 3*HID
#define BB   256
#define TT   512

typedef __attribute__((ext_vector_type(16))) __bf16 v16bf;
typedef __attribute__((ext_vector_type(8)))  float  v8f;

union ABf { v16bf v; unsigned u[8]; };

__device__ __forceinline__ unsigned short f2bf(float f) {   // RNE (weights / h state)
  unsigned u = __builtin_bit_cast(unsigned, f);
  u += 0x7FFFu + ((u >> 16) & 1u);
  return (unsigned short)(u >> 16);
}
__device__ __forceinline__ unsigned pkbf(float a, float b) { // RNE pack (cold paths)
  return (unsigned)f2bf(a) | ((unsigned)f2bf(b) << 16);
}
// single v_perm_b32: packed bf16 {b,a} via truncation (hot activation paths)
__device__ __forceinline__ unsigned pkbf_fast(float a, float b) {
  return __builtin_amdgcn_perm(__builtin_bit_cast(unsigned, b),
                               __builtin_bit_cast(unsigned, a), 0x07060302u);
}
__device__ __forceinline__ float fsigmoid(float x) {
  return 1.f / (1.f + __expf(-x));
}
__device__ __forceinline__ float ftanh(float x) {
  return 2.f / (1.f + __expf(-2.f * x)) - 1.f;   // one v_exp_f32
}

// ---------------------------------------------------------------------------
// Kernel 1: z[b*T+t][0..31] = x[b*T+t][:] @ W_mix^T   (HBM bound: 268 MB of x)
// One workgroup (64 threads / 2 waves) per 16-row M tile; wave w -> N cols
// 16w..16w+15. B tiles stream from global (W_mix is 64 KB, L2-resident).
// ---------------------------------------------------------------------------
__global__ __launch_bounds__(64) void mix_gemm(const float* __restrict__ x,
                                               const float* __restrict__ Wm,
                                               float* __restrict__ z) {
  __shared__ float xs[16 * IN_D];
  const int tid  = threadIdx.x;
  const int wave = tid >> 5;
  const int lane = tid & 31;
  const int nloc = lane & 15;
  const int H    = lane >> 4;
  const size_t mbase = (size_t)blockIdx.x * 16;

  // stage 16 contiguous rows of x (32 KB), fully coalesced
  {
    const float4* src = (const float4*)(x + mbase * IN_D);
    float4* dst = (float4*)xs;
    for (int i = tid; i < (16 * IN_D) / 4; i += 64) dst[i] = src[i];
  }
  __syncthreads();

  const int ncol = wave * 16 + nloc;
  const float* wrow = Wm + (size_t)ncol * IN_D + H * 16;   // lane's half-row of W_mix

  v8f acc = {0.f,0.f,0.f,0.f,0.f,0.f,0.f,0.f};
#pragma unroll 4
  for (int kt = 0; kt < 16; ++kt) {
    ABf A, Bt;
    const float* wr = wrow + kt * 32;
    const float* xr = xs + nloc * IN_D + kt * 32;
#pragma unroll
    for (int d = 0; d < 8; ++d) {
      float2 pb = *(const float2*)(wr + 2 * d);            // B layout: K pairs
      Bt.u[d] = pkbf_fast(pb.x, pb.y);
      const int k = (H * 16) + (d * 2);  // contiguous K layout (matches B)
      float2 pa = *(const float2*)(xr + k);
      A.u[d] = pkbf_fast(pa.x, pa.y);
    }
    acc = __builtin_amdgcn_wmma_f32_16x16x32_bf16(false, A.v, false, Bt.v,
                                                  (short)0, acc, false, false);
  }
#pragma unroll
  for (int d = 0; d < 8; ++d) {
    const int m = d + 8 * H;                  // C layout: vgpr d, lane half H
    z[(mbase + m) * MIXD + ncol] = acc[d];
  }
}

// ---------------------------------------------------------------------------
// Kernel 2: GRU scan + head + tied projection.
// 16 workgroups x 192 threads (6 waves); WG owns batch rows [16*bid, 16*bid+16).
// Wave w owns gate columns 48w..48w+47 (3 WMMA N-tiles). GRU weights live in
// bf16 B-layout registers for all 512 steps (Bih: 24 VGPRs, Bhh: 72 VGPRs).
// ---------------------------------------------------------------------------
__global__ __launch_bounds__(192) void gru_scan(
    const float* __restrict__ z,   const float* __restrict__ Wih,
    const float* __restrict__ Whh, const float* __restrict__ bih,
    const float* __restrict__ bhh, const float* __restrict__ Whd,
    const float* __restrict__ bhd, const float* __restrict__ Wm,
    float* __restrict__ out) {
  __shared__ float    S[16][192];      // r,u pre-activations (cols 0..191)
  __shared__ float    SxN[16][96];     // x-side of n gate (+b_ih)
  __shared__ float    ShN[16][96];     // h-side of n gate (+b_hh)
  __shared__ float    hS[16][96];      // h state, fp32
  __shared__ unsigned hA[3][32][8];    // h packed bf16 in WMMA A-layout (3 K-tiles)
  __shared__ float    zS[16][32];      // staged z_t
  __shared__ float    znS[16][32];     // z_next (epilogue)

  const int tid  = threadIdx.x;
  const int w    = tid >> 5;           // wave 0..5
  const int lane = tid & 31;
  const int nloc = lane & 15;
  const int H    = lane >> 4;
  const int b0   = blockIdx.x * 16;

  // --- preload weights in bf16 B-layout (registers, loop invariant, RNE) ---
  ABf Bih[3];          // gx:  K=32, 3 N-tiles
  ABf Bhh[3][3];       // gh:  [ntile][ktile], K=96 -> 3 K-tiles
  float bI[3], bH[3];
#pragma unroll
  for (int j = 0; j < 3; ++j) {
    const int col = (3 * w + j) * 16 + nloc;   // gate column in [0,288)
    bI[j] = bih[col]; bH[j] = bhh[col];
    const float* wr = Wih + (size_t)col * MIXD + H * 16;
#pragma unroll
    for (int d = 0; d < 8; ++d) {
      float2 p = *(const float2*)(wr + 2 * d);
      Bih[j].u[d] = pkbf(p.x, p.y);
    }
#pragma unroll
    for (int kt = 0; kt < 3; ++kt) {
      const float* hr = Whh + (size_t)col * HID + kt * 32 + H * 16;
#pragma unroll
      for (int d = 0; d < 8; ++d) {
        float2 p = *(const float2*)(hr + 2 * d);
        Bhh[j][kt].u[d] = pkbf(p.x, p.y);
      }
    }
  }

  // h0 = 0
  for (int e = tid; e < 16 * 96; e += 192) ((float*)hS)[e] = 0.f;
  for (int e = tid; e < 3 * 32 * 8; e += 192) ((unsigned*)hA)[e] = 0u;
  __syncthreads();

  unsigned short* hA16 = (unsigned short*)hA;

  for (int t = 0; t < TT; ++t) {
    // --- stage z_t tile (16x32 fp32), coalesced, + prefetch t+1 ---
    if (tid < 128) {
      const int row = tid >> 3, q = tid & 7;
      const float* zp = z + ((size_t)(b0 + row) * TT + t) * MIXD + q * 4;
      *(float4*)&zS[row][q * 4] = *(const float4*)zp;
      if (t + 1 < TT) __builtin_prefetch(zp + MIXD, 0, 3);
    }
    __syncthreads();   // zS ready; previous update phase (hA/hS) complete

    // --- WMMA phase: gx = z_t@Wih^T ; gh = h@Whh^T ---
    ABf Az;
#pragma unroll
    for (int d = 0; d < 8; ++d) {
      const int k = (H * 16) + (d * 2);
      float2 p = *(const float2*)&zS[nloc][k];
      Az.u[d] = pkbf_fast(p.x, p.y);   // 1 v_perm_b32 per dword
    }
    const v8f zero8 = {0.f,0.f,0.f,0.f,0.f,0.f,0.f,0.f};
    v8f ax[3], ah[3];
#pragma unroll
    for (int j = 0; j < 3; ++j) { ax[j] = zero8; ah[j] = zero8; }
#pragma unroll
    for (int j = 0; j < 3; ++j)
      ax[j] = __builtin_amdgcn_wmma_f32_16x16x32_bf16(false, Az.v, false, Bih[j].v,
                                                      (short)0, ax[j], false, false);
#pragma unroll
    for (int kt = 0; kt < 3; ++kt) {
      ABf Ah;
      {  // two ds_load_b128 instead of 8 b32
        uint4 lo = *(const uint4*)&hA[kt][lane][0];
        uint4 hi = *(const uint4*)&hA[kt][lane][4];
        Ah.u[0] = lo.x; Ah.u[1] = lo.y; Ah.u[2] = lo.z; Ah.u[3] = lo.w;
        Ah.u[4] = hi.x; Ah.u[5] = hi.y; Ah.u[6] = hi.z; Ah.u[7] = hi.w;
      }
#pragma unroll
      for (int j = 0; j < 3; ++j)
        ah[j] = __builtin_amdgcn_wmma_f32_16x16x32_bf16(false, Ah.v, false, Bhh[j][kt].v,
                                                        (short)0, ah[j], false, false);
    }
    // write pre-activations to LDS (+biases); n-gate keeps x/h parts separate
    if (w < 4) {
#pragma unroll
      for (int j = 0; j < 3; ++j) {
        const int nl = (3 * w + j) * 16 + nloc;
        const float bsum = bI[j] + bH[j];
#pragma unroll
        for (int d = 0; d < 8; ++d) S[d + 8 * H][nl] = ax[j][d] + ah[j][d] + bsum;
      }
    } else {
#pragma unroll
      for (int j = 0; j < 3; ++j) {
        const int nn = (3 * w + j) * 16 + nloc - 192;
#pragma unroll
        for (int d = 0; d < 8; ++d) {
          SxN[d + 8 * H][nn] = ax[j][d] + bI[j];
          ShN[d + 8 * H][nn] = ah[j][d] + bH[j];
        }
      }
    }
    __syncthreads();   // gate pre-activations visible

    // --- gate/update phase: 16*96 = 1536 elements, 8 per thread ---
#pragma unroll
    for (int p = 0; p < 8; ++p) {
      const int e = tid + 192 * p;
      const int b = e / 96, j = e - b * 96;
      const float r  = fsigmoid(S[b][j]);
      const float u  = fsigmoid(S[b][96 + j]);
      const float n  = ftanh(SxN[b][j] + r * ShN[b][j]);
      const float hn = (1.f - u) * n + u * hS[b][j];
      hS[b][j] = hn;
      // re-pack into bf16 A-layout (RNE) for next step's WMMA
      const int kt = j >> 5, kk = j & 31, g = kk >> 3;
      const int lane_ = b + ((kk >> 4) << 4);
      const int dd = (kk & 15) >> 1;
      hA16[(((kt * 32 + lane_) * 8 + dd) << 1) + (kk & 1)] = f2bf(hn);
    }
    // no barrier here: next iteration's post-stage barrier orders hA/hS reads
  }
  __syncthreads();

  // --- epilogue (tiny FLOPs): z_next = h@W_head^T + b_head ---
  for (int e = tid; e < 16 * MIXD; e += 192) {
    const int b = e >> 5, m = e & 31;
    float a = bhd[m];
    const float* wr = Whd + m * HID;
#pragma unroll 8
    for (int k = 0; k < HID; ++k) a += hS[b][k] * wr[k];
    znS[b][m] = a;
  }
  __syncthreads();

  // y = z_next @ W_mix  (coalesced reads/writes along INPUT_DIM)
  for (int e = tid; e < 16 * IN_D; e += 192) {
    const int b = e >> 9, dcol = e & 511;
    float a = 0.f;
#pragma unroll 8
    for (int m = 0; m < MIXD; ++m) a += znS[b][m] * Wm[m * IN_D + dcol];
    out[(size_t)(b0 + b) * IN_D + dcol] = a;
  }
}

// ---------------------------------------------------------------------------
extern "C" void kernel_launch(void* const* d_in, const int* in_sizes, int n_in,
                              void* d_out, int out_size, void* d_ws, size_t ws_size,
                              hipStream_t stream) {
  const float* x   = (const float*)d_in[0];
  const float* Wm  = (const float*)d_in[1];
  const float* Wih = (const float*)d_in[2];
  const float* Whh = (const float*)d_in[3];
  const float* bih = (const float*)d_in[4];
  const float* bhh = (const float*)d_in[5];
  const float* Whd = (const float*)d_in[6];
  const float* bhd = (const float*)d_in[7];
  float* out = (float*)d_out;
  float* z   = (float*)d_ws;                 // B*T*MIX fp32 = 16.8 MB scratch

  mix_gemm<<<(BB * TT) / 16, 64, 0, stream>>>(x, Wm, z);
  gru_scan<<<BB / 16, 192, 0, stream>>>(z, Wih, Whh, bih, bhh, Whd, bhd, Wm, out);
}
